// MutualAttention_1975684956761
// MI455X (gfx1250) — compile-verified, matrix operand order checked by reading, not run on hardware
//
#include <hip/hip_runtime.h>
#include <hip/hip_fp16.h>
#include <math.h>

// ---------------------------------------------------------------------------
// MutualAttention for MI455X (gfx1250, wave32, WMMA).
// - All matmuls: V_WMMA_F32_16X16X32_F16 (f16 A/B, f32 accumulate).
// - Flash-style attention: [B,H,N,N] scores (2.1 GB) never touch HBM.
// - Fused Q/K/V projections (X read once), double-buffered K/V staging with
//   async global->LDS (ASYNCcnt path), GEMM+LayerNorm fusion.
// ---------------------------------------------------------------------------

typedef __attribute__((ext_vector_type(16))) _Float16 v16h;
typedef __attribute__((ext_vector_type(8)))  _Float16 h8;
typedef __attribute__((ext_vector_type(8)))  float    v8f;
typedef __attribute__((ext_vector_type(4)))  int      i4;

#define C_DIM   256
#define N_SEQ   4096
#define B_SZ    4
#define H_HEADS 4
#define D_HEAD  64
#define M_ROWS  (B_SZ * N_SEQ)   // 16384
#define SCALE_F 0.125f           // D^-0.5 = 64^-0.5

// Async global->LDS (ASYNCcnt path) if this toolchain exposes the builtins.
// Probe round confirmed: 4 args, (AS1 v4i* src, AS3 v4i* dst, imm, imm).
#if defined(__has_builtin)
#if __has_builtin(__builtin_amdgcn_global_load_async_to_lds_b128) && \
    __has_builtin(__builtin_amdgcn_s_wait_asynccnt)
#define HAVE_ASYNC_LDS 1
#endif
#endif
#ifndef HAVE_ASYNC_LDS
#define HAVE_ASYNC_LDS 0
#endif

#if HAVE_ASYNC_LDS
typedef __attribute__((address_space(1))) i4 as1_i4;
typedef __attribute__((address_space(3))) i4 as3_i4;
#endif

// Load a 16x32 f16 WMMA fragment (A or B) from row-major [16][ld] f16 storage.
// ISA layout (CDNA5 ISA): lane L holds row (L&15); lanes 0-15 carry
// K=0..7 (VGPR0-3) and K=16..23 (VGPR4-7); lanes 16-31 carry K=8..15 / 24..31.
__device__ __forceinline__ v16h load_frag_k32(const _Float16* base, int ld, int lane) {
  const int r  = lane & 15;
  const int k0 = (lane >> 4) << 3;          // 0 or 8
  const _Float16* p = base + r * ld + k0;
  union { v16h v; h8 h[2]; } u;
  u.h[0] = *(const h8*)(p);
  u.h[1] = *(const h8*)(p + 16);
  return u.v;
}

// ---------------------------------------------------------------------------
// Kernel 1: fused triple projection.  For one input X_f32[M,256], compute
// Yw_f16 = X @ Ww^T + bw for w=0..2, staging X once per K-step (12 WMMAs
// per staged X tile).  WG = 256 thr (8 waves), tile 128(M) x 64(N).
// ---------------------------------------------------------------------------
__global__ __launch_bounds__(256)
void proj3_gemm_kernel(const float* __restrict__ X,
                       const float* __restrict__ W0, const float* __restrict__ b0,
                       _Float16* __restrict__ Y0,
                       const float* __restrict__ W1, const float* __restrict__ b1,
                       _Float16* __restrict__ Y1,
                       const float* __restrict__ W2, const float* __restrict__ b2,
                       _Float16* __restrict__ Y2) {
  __shared__ alignas(16) _Float16 lx[128 * 32];       // X tile (f16)
  __shared__ alignas(16) _Float16 lw[3][64 * 32];     // 3 weight tiles, [n][k]
  const int tid = threadIdx.x, wave = tid >> 5, lane = tid & 31;
  const int m0 = blockIdx.x * 128, n0 = blockIdx.y * 64;
  const float* Ws[3] = {W0, W1, W2};
  const float* bs[3] = {b0, b1, b2};
  _Float16*    Ys[3] = {Y0, Y1, Y2};

  v8f acc[3][4] = {};
  for (int k0 = 0; k0 < C_DIM; k0 += 32) {
    // Stage X: 128x32 f32 -> f16 (1024 float4 / 256 threads = 4 each)
#pragma unroll
    for (int i = 0; i < 4; i++) {
      int idx = tid + i * 256, r = idx >> 3, cs = idx & 7;
      const float* src = X + (size_t)(m0 + r) * C_DIM + k0 + cs * 4;
      float4 v = *(const float4*)src;
      if (k0 + 32 < C_DIM) __builtin_prefetch(src + 32, 0, 3);
      _Float16* d = lx + r * 32 + cs * 4;
      d[0] = (_Float16)v.x; d[1] = (_Float16)v.y;
      d[2] = (_Float16)v.z; d[3] = (_Float16)v.w;
    }
    // Stage 3 weight tiles (B[k][n] = W[n][k]: rows already K-contiguous)
#pragma unroll
    for (int w = 0; w < 3; w++)
#pragma unroll
      for (int i = 0; i < 2; i++) {
        int idx = tid + i * 256, r = idx >> 3, cs = idx & 7;
        float4 v = *(const float4*)(Ws[w] + (size_t)(n0 + r) * C_DIM + k0 + cs * 4);
        _Float16* d = lw[w] + r * 32 + cs * 4;
        d[0] = (_Float16)v.x; d[1] = (_Float16)v.y;
        d[2] = (_Float16)v.z; d[3] = (_Float16)v.w;
      }
    __syncthreads();
    v16h a = load_frag_k32(lx + wave * 16 * 32, 32, lane);
#pragma unroll
    for (int w = 0; w < 3; w++)
#pragma unroll
      for (int t = 0; t < 4; t++) {
        v16h b = load_frag_k32(lw[w] + t * 16 * 32, 32, lane);
        acc[w][t] = __builtin_amdgcn_wmma_f32_16x16x32_f16(
            false, a, false, b, (short)0, acc[w][t], false, false);
      }
    __syncthreads();
  }
  // C/D layout: VGPR j -> row j + 8*(lane>=16); col = lane&15 within tile
#pragma unroll
  for (int w = 0; w < 3; w++)
#pragma unroll
    for (int t = 0; t < 4; t++) {
      int col = n0 + t * 16 + (lane & 15);
      float bv = bs[w][col];
#pragma unroll
      for (int j = 0; j < 8; j++) {
        int row = m0 + wave * 16 + ((lane >> 4) << 3) + j;
        Ys[w][(size_t)row * C_DIM + col] = (_Float16)(acc[w][t][j] + bv);
      }
    }
}

// ---------------------------------------------------------------------------
// Kernel 2: flash attention, one (b, h, 128-row block) per WG.
// Q/K/V/O are f16 [B,N,C] with head h at column offset h*64 (stride C).
// Each wave owns 16 Q rows; double-buffered 64-col K/V blocks in LDS.
// K staged via async global->LDS (ASYNCcnt); V is transposed on the way into
// LDS so PV B-fragments are K-contiguous.
// ---------------------------------------------------------------------------
__global__ __launch_bounds__(256)
void attn_kernel(const _Float16* __restrict__ Q, const _Float16* __restrict__ K,
                 const _Float16* __restrict__ V, _Float16* __restrict__ O) {
  __shared__ alignas(16) _Float16 kt[2][64 * 64];     // K blocks, [row][d]
  __shared__ alignas(16) _Float16 vt[2][64 * 64];     // V blocks transposed [n][k]
  __shared__ alignas(16) _Float16 pt[8][16 * 64];     // per-wave P bounce
  const int tid = threadIdx.x, wave = tid >> 5, lane = tid & 31;
  const int b = blockIdx.z, h = blockIdx.y;
  const int r0 = blockIdx.x * 128 + wave * 16;
  const size_t base = (size_t)b * N_SEQ * C_DIM + (size_t)h * D_HEAD;
  const _Float16* Qb = Q + base;
  const _Float16* Kb = K + base;
  const _Float16* Vb = V + base;
  _Float16* Ob = O + base;

  // Persistent Q fragments: 16 rows x 64 (D) = two 16x32 A-fragments
  v16h aq[2];
  {
    const int r = lane & 15, k0 = (lane >> 4) << 3;
    const _Float16* p = Qb + (size_t)(r0 + r) * C_DIM;
#pragma unroll
    for (int c = 0; c < 2; c++) {
      union { v16h v; h8 hh[2]; } u;
      u.hh[0] = *(const h8*)(p + c * 32 + k0);
      u.hh[1] = *(const h8*)(p + c * 32 + k0 + 16);
      aq[c] = u.v;
    }
  }

  auto stage = [&](int buf, int jb) {
#pragma unroll
    for (int i = 0; i < 2; i++) {
      int idx = tid + i * 256, r = idx >> 3, seg = idx & 7;
      const _Float16* kp = Kb + (size_t)(jb + r) * C_DIM + seg * 8;
      _Float16* kd = &kt[buf][r * 64 + seg * 8];
#if HAVE_ASYNC_LDS
      __builtin_amdgcn_global_load_async_to_lds_b128(
          (as1_i4*)(uintptr_t)kp, (as3_i4*)(uintptr_t)kd, 0, 0);
#else
      *(h8*)kd = *(const h8*)kp;
#endif
      h8 vv = *(const h8*)(Vb + (size_t)(jb + r) * C_DIM + seg * 8);
#pragma unroll
      for (int c = 0; c < 8; c++) vt[buf][(seg * 8 + c) * 64 + r] = vv[c];
    }
  };

  float mrow[8], lrow[8];
#pragma unroll
  for (int j = 0; j < 8; j++) { mrow[j] = -1e30f; lrow[j] = 0.f; }
  v8f oacc[4] = {};

  stage(0, 0);
  for (int it = 0; it < N_SEQ / 64; ++it) {
    const int cur = it & 1;
#if HAVE_ASYNC_LDS
    __builtin_amdgcn_s_wait_asynccnt(0);   // my async K tile landed in LDS
#endif
    __syncthreads();                        // publish cur; nxt buffer free
    if (it + 1 < N_SEQ / 64) stage(cur ^ 1, (it + 1) * 64);

    // S(16x64) = Q(16x64) · K^T  — B[k][n] = K[n][k] (kt is K-contiguous)
    v8f s[4] = {};
#pragma unroll
    for (int t = 0; t < 4; t++)
#pragma unroll
      for (int c = 0; c < 2; c++) {
        v16h bf = load_frag_k32(&kt[cur][t * 16 * 64 + c * 32], 64, lane);
        s[t] = __builtin_amdgcn_wmma_f32_16x16x32_f16(
            false, aq[c], false, bf, (short)0, s[t], false, false);
      }

    // Online softmax. Row M lives across 16 lanes at fixed VGPR index j.
    float mnew[8], rs[8], fac[8];
#pragma unroll
    for (int j = 0; j < 8; j++) {
      float mx = mrow[j];
#pragma unroll
      for (int t = 0; t < 4; t++) mx = fmaxf(mx, s[t][j] * SCALE_F);
#pragma unroll
      for (int o = 1; o < 16; o <<= 1) mx = fmaxf(mx, __shfl_xor(mx, o, 32));
      mnew[j] = mx; rs[j] = 0.f;
    }
#pragma unroll
    for (int t = 0; t < 4; t++)
#pragma unroll
      for (int j = 0; j < 8; j++) {
        float p = __expf(s[t][j] * SCALE_F - mnew[j]);
        s[t][j] = p; rs[j] += p;
      }
#pragma unroll
    for (int j = 0; j < 8; j++) {
#pragma unroll
      for (int o = 1; o < 16; o <<= 1) rs[j] += __shfl_xor(rs[j], o, 32);
      fac[j] = __expf(mrow[j] - mnew[j]);
      lrow[j] = lrow[j] * fac[j] + rs[j];
      mrow[j] = mnew[j];
    }
#pragma unroll
    for (int t = 0; t < 4; t++)
#pragma unroll
      for (int j = 0; j < 8; j++) oacc[t][j] *= fac[j];

    // Bounce P through per-wave LDS (accumulator layout -> A-fragment layout)
    _Float16* pw = pt[wave];
#pragma unroll
    for (int t = 0; t < 4; t++)
#pragma unroll
      for (int j = 0; j < 8; j++)
        pw[(((lane >> 4) << 3) + j) * 64 + t * 16 + (lane & 15)] =
            (_Float16)s[t][j];

    // O(16x64) += P(16x64) · V(64x64)  — vt is [n][k], K-contiguous
#pragma unroll
    for (int c = 0; c < 2; c++) {
      v16h ap = load_frag_k32(pw + c * 32, 64, lane);
#pragma unroll
      for (int t = 0; t < 4; t++) {
        v16h bf = load_frag_k32(&vt[cur][t * 16 * 64 + c * 32], 64, lane);
        oacc[t] = __builtin_amdgcn_wmma_f32_16x16x32_f16(
            false, ap, false, bf, (short)0, oacc[t], false, false);
      }
    }
  }

  // Normalize and store (merged-head layout: col h*64 + d of [B,N,C])
#pragma unroll
  for (int t = 0; t < 4; t++) {
    int col = t * 16 + (lane & 15);
#pragma unroll
    for (int j = 0; j < 8; j++) {
      int row = r0 + ((lane >> 4) << 3) + j;
      Ob[(size_t)row * C_DIM + col] = (_Float16)(oacc[t][j] / lrow[j]);
    }
  }
}

// ---------------------------------------------------------------------------
// Kernel 3: fused = concat(o1,o2)[M,512] @ Wp[256,512]^T + bp, then LayerNorm
// over C=256, written straight to the output (no fp32 round-trip buffer).
// Wave tile = 16 rows x full 256 cols (16 accumulators) so each wave owns
// complete rows; LN reduces in-register + 16-lane shfl_xor.
// ---------------------------------------------------------------------------
__global__ __launch_bounds__(256)
void fused_gemm_ln_kernel(const _Float16* __restrict__ X1, const _Float16* __restrict__ X2,
                          const float* __restrict__ W, const float* __restrict__ bias,
                          const float* __restrict__ gamma, const float* __restrict__ beta,
                          float* __restrict__ Y) {
  __shared__ alignas(16) _Float16 lx[128 * 32];     // 8 KB
  __shared__ alignas(16) _Float16 lw[256 * 32];     // 16 KB, [n][k]
  const int tid = threadIdx.x, wave = tid >> 5, lane = tid & 31;
  const int m0 = blockIdx.x * 128;

  v8f acc[16] = {};
  for (int k0 = 0; k0 < 512; k0 += 32) {
    const _Float16* X = (k0 < 256) ? X1 : X2;
    const int kk = k0 & 255;
    // Stage X (f16): 128x32 = 512 h8 / 256 thr = 2 each
#pragma unroll
    for (int i = 0; i < 2; i++) {
      int idx = tid + i * 256, r = idx >> 2, seg = idx & 3;
      *(h8*)(lx + r * 32 + seg * 8) =
          *(const h8*)(X + (size_t)(m0 + r) * C_DIM + kk + seg * 8);
    }
    // Stage Wp tile: 256(n) x 32(k) f32 -> f16 (2048 float4 / 256 = 8 each)
#pragma unroll
    for (int i = 0; i < 8; i++) {
      int idx = tid + i * 256, r = idx >> 3, cs = idx & 7;
      float4 v = *(const float4*)(W + (size_t)r * 512 + k0 + cs * 4);
      _Float16* d = lw + r * 32 + cs * 4;
      d[0] = (_Float16)v.x; d[1] = (_Float16)v.y;
      d[2] = (_Float16)v.z; d[3] = (_Float16)v.w;
    }
    __syncthreads();
    v16h a = load_frag_k32(lx + wave * 16 * 32, 32, lane);
#pragma unroll
    for (int t = 0; t < 16; t++) {
      v16h b = load_frag_k32(lw + t * 16 * 32, 32, lane);
      acc[t] = __builtin_amdgcn_wmma_f32_16x16x32_f16(
          false, a, false, b, (short)0, acc[t], false, false);
    }
    __syncthreads();
  }

  // Bias
#pragma unroll
  for (int t = 0; t < 16; t++) {
    float bv = bias[t * 16 + (lane & 15)];
#pragma unroll
    for (int j = 0; j < 8; j++) acc[t][j] += bv;
  }
  // LayerNorm stats per row: lane j-partial over its 16 tiles, then 16-lane
  // shfl reduction => full 256-element sum / sum-of-squares.
  float mu[8], rv[8];
#pragma unroll
  for (int j = 0; j < 8; j++) {
    float s1 = 0.f, s2 = 0.f;
#pragma unroll
    for (int t = 0; t < 16; t++) { float v = acc[t][j]; s1 += v; s2 += v * v; }
#pragma unroll
    for (int o = 1; o < 16; o <<= 1) {
      s1 += __shfl_xor(s1, o, 32);
      s2 += __shfl_xor(s2, o, 32);
    }
    float m = s1 * (1.0f / C_DIM);
    mu[j] = m;
    rv[j] = rsqrtf(s2 * (1.0f / C_DIM) - m * m + 1e-5f);
  }
#pragma unroll
  for (int t = 0; t < 16; t++) {
    int col = t * 16 + (lane & 15);
    float g = gamma[col], bb = beta[col];
#pragma unroll
    for (int j = 0; j < 8; j++) {
      int row = m0 + wave * 16 + ((lane >> 4) << 3) + j;
      Y[(size_t)row * C_DIM + col] = (acc[t][j] - mu[j]) * rv[j] * g + bb;
    }
  }
}

// ---------------------------------------------------------------------------
extern "C" void kernel_launch(void* const* d_in, const int* in_sizes, int n_in,
                              void* d_out, int out_size, void* d_ws, size_t ws_size,
                              hipStream_t stream) {
  (void)in_sizes; (void)n_in; (void)out_size; (void)ws_size;
  const float* f_freq = (const float*)d_in[0];
  const float* f_spat = (const float*)d_in[1];
  const float* Wqf = (const float*)d_in[2];  const float* bqf = (const float*)d_in[3];
  const float* Wks = (const float*)d_in[4];  const float* bks = (const float*)d_in[5];
  const float* Wvs = (const float*)d_in[6];  const float* bvs = (const float*)d_in[7];
  const float* Wqs = (const float*)d_in[8];  const float* bqs = (const float*)d_in[9];
  const float* Wkf = (const float*)d_in[10]; const float* bkf = (const float*)d_in[11];
  const float* Wvf = (const float*)d_in[12]; const float* bvf = (const float*)d_in[13];
  const float* Wp  = (const float*)d_in[14]; const float* bp  = (const float*)d_in[15];
  const float* gamma = (const float*)d_in[16];
  const float* beta  = (const float*)d_in[17];
  float* out = (float*)d_out;

  char* ws = (char*)d_ws;
  const size_t HBUF = (size_t)M_ROWS * C_DIM * sizeof(_Float16);  // 8 MB
  _Float16* qf = (_Float16*)(ws + 0 * HBUF);
  _Float16* ks = (_Float16*)(ws + 1 * HBUF);
  _Float16* vs = (_Float16*)(ws + 2 * HBUF);
  _Float16* qs = (_Float16*)(ws + 3 * HBUF);
  _Float16* kf = (_Float16*)(ws + 4 * HBUF);
  _Float16* vf = (_Float16*)(ws + 5 * HBUF);
  _Float16* o1 = (_Float16*)(ws + 6 * HBUF);
  _Float16* o2 = (_Float16*)(ws + 7 * HBUF);

  dim3 gp(M_ROWS / 128, C_DIM / 64);
  // f_freq -> Qf, Kf, Vf   (X staged once for all three)
  proj3_gemm_kernel<<<gp, 256, 0, stream>>>(f_freq, Wqf, bqf, qf,
                                            Wkf, bkf, kf, Wvf, bvf, vf);
  // f_spat -> Qs, Ks, Vs
  proj3_gemm_kernel<<<gp, 256, 0, stream>>>(f_spat, Wqs, bqs, qs,
                                            Wks, bks, ks, Wvs, bvs, vs);

  dim3 ga(N_SEQ / 128, H_HEADS, B_SZ);
  attn_kernel<<<ga, 256, 0, stream>>>(qf, ks, vs, o1);  // freq -> spat
  attn_kernel<<<ga, 256, 0, stream>>>(qs, kf, vf, o2);  // spat -> freq

  fused_gemm_ln_kernel<<<dim3(M_ROWS / 128), 256, 0, stream>>>(
      o1, o2, Wp, bp, gamma, beta, out);
}
